// MSA2Pair_60790967108022
// MI455X (gfx1250) — matrix operand order repaired and checked by reading, not compiled, not run on hardware
//
#include <hip/hip_runtime.h>
#include <hip/hip_bf16.h>

typedef _Float16 half_t;
typedef __attribute__((ext_vector_type(8)))  _Float16 v8h;
typedef __attribute__((ext_vector_type(16))) _Float16 v16h;
typedef __attribute__((ext_vector_type(8)))  float    v8f;

// ---------------------------------------------------------------------------
// WMMA helpers (CDNA5 v_wmma_f32_16x16x32_f16, wave32)
// A-matrix 16x32 f16 layout: lane L: M = L&15, K chunks [hb*8..hb*8+7] and
// [16+hb*8 .. 16+hb*8+7] where hb = L>>4.  (ISA 7.12.2)
// B-matrix 32x16 f16 layout: lane L: N = L&15, K chunks [hb*8..hb*8+7] and
// [16+hb*8 .. 16+hb*8+7] (same as A).
// C/D 16x16 f32: lane L: N = L&15, VGPR j holds row hb*8+j.
// ---------------------------------------------------------------------------
__device__ __forceinline__ v8f wmma16(v16h a, v16h b, v8f c) {
  return __builtin_amdgcn_wmma_f32_16x16x32_f16(
      /*neg_a*/false, a, /*neg_b*/false, b, /*c_mod*/(short)0, c,
      /*reuse_a*/false, /*reuse_b*/false);
}

// Load A fragment from [M][K] row-major (ld = leading dim in elements).
__device__ __forceinline__ v16h frag_a(const half_t* base, int ld) {
  int lane = threadIdx.x & 31;
  int m = lane & 15, hb = lane >> 4;
  const half_t* r = base + (size_t)m * ld + hb * 8;
  v8h lo = *(const v8h*)(r);
  v8h hi = *(const v8h*)(r + 16);
  v16h f;
#pragma unroll
  for (int t = 0; t < 8; ++t) { f[t] = lo[t]; f[8 + t] = hi[t]; }
  return f;
}

// Load B fragment from B-transposed, i.e. [N][K] row-major (ld in elements).
__device__ __forceinline__ v16h frag_b(const half_t* bt, int ld) {
  int lane = threadIdx.x & 31;
  int n = lane & 15, hb = lane >> 4;
  const half_t* r = bt + (size_t)n * ld + hb * 8;
  v8h lo = *(const v8h*)(r);
  v8h hi = *(const v8h*)(r + 16);
  v16h f;
#pragma unroll
  for (int t = 0; t < 8; ++t) { f[t] = lo[t]; f[8 + t] = hi[t]; }
  return f;
}

// LDS byte-address of a generic pointer that aliases __shared__ memory
// (for async global->LDS ops: VDST VGPR holds the wave-relative LDS address).
__device__ __forceinline__ unsigned lds_addr_of(const void* p) {
  return (unsigned)(unsigned long long)(__attribute__((address_space(3))) const void*)p;
}

// ---------------------------------------------------------------------------
// Problem constants
// ---------------------------------------------------------------------------
#define LSEQ 256
#define NMSA 64
#define NF   64
#define NPJ  32
#define NFO  128

// ---------------------------------------------------------------------------
// Weight preprocessing: f32 -> f16 copies / tap-transposes
//   W2h   [128][1024]
//   Wcath [128][256]      (Wc0 columns 0..255: pair_orig ++ pair2)
//   W3t*  [9][128][128]   (tap-major, [o][c] row-major per tap)
// ---------------------------------------------------------------------------
__global__ void k_prep(const float* __restrict__ W2, const float* __restrict__ Wc0,
                       const float* __restrict__ Wr1, const float* __restrict__ Wr2,
                       half_t* W2h, half_t* Wcath, half_t* W3t1, half_t* W3t2) {
  int idx = blockIdx.x * 256 + threadIdx.x;
  if (idx < 128 * 1024) W2h[idx] = (half_t)W2[idx];
  if (idx < 128 * 256) {
    int o = idx >> 8, c = idx & 255;
    Wcath[idx] = (half_t)Wc0[o * 512 + c];
  }
  if (idx < 9 * 128 * 128) {
    int tap = idx / 16384, rem = idx % 16384;
    int o = rem >> 7, c = rem & 127;
    size_t s = ((size_t)(o * 128 + c)) * 9 + tap;   // Wr [O][I][3][3]
    W3t1[idx] = (half_t)Wr1[s];
    W3t2[idx] = (half_t)Wr2[s];
  }
}

// ---------------------------------------------------------------------------
// proj_1: xdT[i*32+p][n] = (msa[n,i,:]·W1[p,:] + b1[p]) * 0.125
// (0.125^2 = 1/64 = outer-product mean divisor, split into both operands)
// ---------------------------------------------------------------------------
__global__ void k_xdown(const float* __restrict__ msa, const float* __restrict__ W1,
                        const float* __restrict__ b1, half_t* __restrict__ xdT) {
  int idx = blockIdx.x * 256 + threadIdx.x;          // 8192*64
  int m = idx >> 6, n = idx & 63;
  int i = m >> 5, p = m & 31;
  const float* mr = msa + ((size_t)(n * LSEQ + i)) * NF;
  const float* wr = W1 + p * NF;
  float s = b1[p];
#pragma unroll 4
  for (int f = 0; f < NF; ++f) s += mr[f] * wr[f];
  xdT[(size_t)m * 64 + n] = (half_t)(0.125f * s);
}

// feat[l][c] = c<64 ? mean_n msa[n,l,c] : msa[0,l,c-64]
__global__ void k_feat(const float* __restrict__ msa, float* __restrict__ feat) {
  int idx = blockIdx.x * 256 + threadIdx.x;          // 256*128
  int l = idx >> 7, c = idx & 127;
  float v;
  if (c < 64) {
    float s = 0.f;
    for (int n = 0; n < NMSA; ++n) s += msa[((size_t)(n * LSEQ + l)) * NF + c];
    v = s * (1.f / 64.f);
  } else {
    v = msa[(size_t)l * NF + (c - 64)];
  }
  feat[idx] = v;
}

// Rank-1 broadcast terms of the 1x1 conv:
//   Lc[i][o] = feat[i]·Wc0[o,256:384],  Rc[l][o] = feat[l]·Wc0[o,384:512]
__global__ void k_lr(const float* __restrict__ feat, const float* __restrict__ Wc0,
                     float* __restrict__ Lc, float* __restrict__ Rc) {
  int idx = blockIdx.x * 256 + threadIdx.x;          // 256*128
  int pos = idx >> 7, o = idx & 127;
  const float* fr = feat + pos * 128;
  const float* wl = Wc0 + o * 512 + 256;
  const float* wr = Wc0 + o * 512 + 384;
  float sl = 0.f, sr = 0.f;
#pragma unroll 4
  for (int c = 0; c < 128; ++c) { sl += fr[c] * wl[c]; sr += fr[c] * wr[c]; }
  Lc[idx] = sl; Rc[idx] = sr;
}

// pair_orig f32 -> cat[pix][0..127] f16   (cat = [65536][256] f16)
__global__ void k_cvt_po(const float* __restrict__ po, half_t* __restrict__ cat) {
  size_t idx = (size_t)blockIdx.x * 256 + threadIdx.x;   // 65536*128
  size_t pix = idx >> 7; int c = (int)(idx & 127);
  cat[pix * 256 + c] = (half_t)po[idx];
}

// ---------------------------------------------------------------------------
// Fused: outer-product GEMM (C = XdT * Xd, per-pixel 32x32 block)
//        -> per-pixel LayerNorm (cross-lane) -> proj_2 (x W2 + b2)
// WG = 256 threads (8 waves) handles 32 pixels (fixed i, 32 consecutive l).
// Stage A: 4 pixels/wave, 2x2 WMMA tiles x K=64 (8 WMMAs/pixel).
// Stage B: normalized y staged in LDS as A[32 px][1024] (64 KB); each wave
//          owns a 16-wide o-block; one B-fragment feeds both M-tiles
//          (halves W2 traffic), K=1024 -> 64 WMMAs/wave.
// ---------------------------------------------------------------------------
__global__ void __launch_bounds__(256)
k_pair(const half_t* __restrict__ xdT, const float* __restrict__ g_ln,
       const float* __restrict__ beta_ln, const half_t* __restrict__ W2h,
       const float* __restrict__ b2, half_t* __restrict__ cat) {
  __shared__ __align__(16) half_t Abuf[32 * 1024];   // 64 KB
  const int i = blockIdx.y;       // 0..255
  const int lb = blockIdx.x;      // 0..7
  const int tid = threadIdx.x;
  const int w = tid >> 5, lane = tid & 31;
  const int m16 = lane & 15, hb = lane >> 4;

  // ----- Stage A: outer product + LN -----
  for (int pxi = 0; pxi < 4; ++pxi) {
    const int px = w * 4 + pxi;
    const int l = lb * 32 + px;
    v8f acc[4] = {v8f{}, v8f{}, v8f{}, v8f{}};   // [ti][tl]
#pragma unroll
    for (int ks = 0; ks < 2; ++ks) {
      v16h a0 = frag_a(xdT + (size_t)(i * 32 + 0) * 64 + ks * 32, 64);
      v16h a1 = frag_a(xdT + (size_t)(i * 32 + 16) * 64 + ks * 32, 64);
      v16h b0 = frag_b(xdT + (size_t)(l * 32 + 0) * 64 + ks * 32, 64);
      v16h b1 = frag_b(xdT + (size_t)(l * 32 + 16) * 64 + ks * 32, 64);
      acc[0] = wmma16(a0, b0, acc[0]);
      acc[1] = wmma16(a0, b1, acc[1]);
      acc[2] = wmma16(a1, b0, acc[2]);
      acc[3] = wmma16(a1, b1, acc[3]);
    }
    // LayerNorm over the full 32x32 block (all 1024 values in this wave)
    float s1 = 0.f, s2 = 0.f;
#pragma unroll
    for (int t = 0; t < 4; ++t)
#pragma unroll
      for (int j = 0; j < 8; ++j) { float v = acc[t][j]; s1 += v; s2 += v * v; }
#pragma unroll
    for (int off = 16; off >= 1; off >>= 1) {
      s1 += __shfl_xor(s1, off, 32);
      s2 += __shfl_xor(s2, off, 32);
    }
    const float mu = s1 * (1.f / 1024.f);
    const float var = s2 * (1.f / 1024.f) - mu * mu;
    const float rs = rsqrtf(var + 1e-5f);
#pragma unroll
    for (int t = 0; t < 4; ++t) {
      const int ti = t >> 1, tl = t & 1;
#pragma unroll
      for (int j = 0; j < 8; ++j) {
        const int p = ti * 16 + hb * 8 + j;
        const int q = tl * 16 + m16;
        const int pq = p * 32 + q;
        const float y = (acc[t][j] - mu) * rs * g_ln[pq] + beta_ln[pq];
        Abuf[px * 1024 + pq] = (half_t)y;
      }
    }
  }
  __syncthreads();

  // ----- Stage B: proj_2, o-block per wave, 2 M-tiles per B-fragment -----
  const int obase = w * 16;
  v8f acc0 = v8f{}, acc1 = v8f{};
  for (int kk = 0; kk < 32; ++kk) {
    v16h b = frag_b(W2h + (size_t)obase * 1024 + kk * 32, 1024);
    v16h a0 = frag_a(Abuf + kk * 32, 1024);
    v16h a1 = frag_a(Abuf + 16 * 1024 + kk * 32, 1024);
    acc0 = wmma16(a0, b, acc0);
    acc1 = wmma16(a1, b, acc1);
  }
  const int o = obase + m16;
  const float bias = b2[o];
#pragma unroll
  for (int j = 0; j < 8; ++j) {
    const int px0 = hb * 8 + j;
    const int l0 = lb * 32 + px0;
    cat[(size_t)(i * LSEQ + l0) * 256 + 128 + o] = (half_t)(acc0[j] + bias);
    const int l1 = l0 + 16;
    cat[(size_t)(i * LSEQ + l1) * 256 + 128 + o] = (half_t)(acc1[j] + bias);
  }
}

// ---------------------------------------------------------------------------
// 1x1 conv (K=256 GEMM over cat) + broadcast Lc/Rc -> pre-norm f32 NHWC
// ---------------------------------------------------------------------------
__global__ void __launch_bounds__(256)
k_conv1(const half_t* __restrict__ cat, const half_t* __restrict__ Wcath,
        const float* __restrict__ Lc, const float* __restrict__ Rc,
        float* __restrict__ out) {
  const int i = blockIdx.y, lb = blockIdx.x, tid = threadIdx.x;
  const int w = tid >> 5, lane = tid & 31, m16 = lane & 15, hb = lane >> 4;
  const int obase = w * 16;
  v8f acc = v8f{};
  const half_t* abase = cat + (size_t)(i * LSEQ + lb * 16) * 256;
#pragma unroll
  for (int kk = 0; kk < 8; ++kk) {
    v16h a = frag_a(abase + kk * 32, 256);
    v16h b = frag_b(Wcath + (size_t)obase * 256 + kk * 32, 256);
    acc = wmma16(a, b, acc);
  }
  const int o = obase + m16;
  const float lci = Lc[i * 128 + o];
#pragma unroll
  for (int j = 0; j < 8; ++j) {
    const int l = lb * 16 + hb * 8 + j;
    out[(size_t)(i * LSEQ + l) * 128 + o] = acc[j] + lci + Rc[l * 128 + o];
  }
}

// ---------------------------------------------------------------------------
// 3x3 SAME conv as implicit GEMM, 8 output rows x 16 cols per WG.
// Halo tile [10][18][128] f16 (45 KB) filled with GLOBAL_LOAD_ASYNC_TO_LDS
// (ASYNCcnt-tracked; zero-padding border via plain LDS stores), then
// 9 taps x 4 k-steps: one B-fragment reused across 8 A-rows -> 288 WMMAs/wave.
// ---------------------------------------------------------------------------
__global__ void __launch_bounds__(256)
k_conv3(const half_t* __restrict__ xin, const half_t* __restrict__ wt,
        float* __restrict__ out) {
  __shared__ __align__(16) half_t tile[10 * 18 * 128];   // 46080 B
  const int ib = blockIdx.y;      // 0..31 (8-row block)
  const int lb = blockIdx.x;      // 0..15
  const int i0 = ib * 8;
  const int tid = threadIdx.x;

  // Halo fill: 10*18*8 = 1440 chunks of 16 halves (32 B each).
  for (int c = tid; c < 10 * 18 * 8; c += 256) {
    const int row = c / 144;          // 0..9
    const int rem = c % 144;
    const int col = rem / 8;          // 0..17
    const int ck = rem % 8;
    const int gi = i0 + row - 1;
    const int gl = lb * 16 + col - 1;
    half_t* dst = &tile[(row * 18 + col) * 128 + ck * 16];
    if (gi >= 0 && gi < LSEQ && gl >= 0 && gl < LSEQ) {
      const half_t* src = xin + ((size_t)(gi * LSEQ + gl) * 128 + ck * 16);
      // async global->LDS copy, tracked by ASYNCcnt (CDNA5)
      asm volatile("global_load_async_to_lds_b128 %0, %1, off"
                   :: "v"(lds_addr_of(dst)), "v"(src) : "memory");
      asm volatile("global_load_async_to_lds_b128 %0, %1, off"
                   :: "v"(lds_addr_of(dst + 8)), "v"(src + 8) : "memory");
    } else {
      v8h z = v8h{};
      *(v8h*)(dst) = z;
      *(v8h*)(dst + 8) = z;
    }
  }
  asm volatile("s_wait_asynccnt 0x0" ::: "memory");
  __syncthreads();

  const int w = tid >> 5, lane = tid & 31, m16 = lane & 15, hb = lane >> 4;
  const int obase = w * 16;
  v8f acc[8] = {v8f{}, v8f{}, v8f{}, v8f{}, v8f{}, v8f{}, v8f{}, v8f{}};
#pragma unroll
  for (int dy = 0; dy < 3; ++dy) {
#pragma unroll
    for (int dx = 0; dx < 3; ++dx) {
      const half_t* wb = wt + (size_t)((dy * 3 + dx) * 128 + obase) * 128;
#pragma unroll
      for (int kk = 0; kk < 4; ++kk) {
        v16h b = frag_b(wb + kk * 32, 128);
#pragma unroll
        for (int r = 0; r < 8; ++r) {
          // lane m -> tile col m+dx, tile row r+dy
          v16h a = frag_a(&tile[((r + dy) * 18 + dx) * 128 + kk * 32], 128);
          acc[r] = wmma16(a, b, acc[r]);
        }
      }
    }
  }
  const int o = obase + m16;
#pragma unroll
  for (int r = 0; r < 8; ++r) {
#pragma unroll
    for (int j = 0; j < 8; ++j) {
      const int l = lb * 16 + hb * 8 + j;
      out[(size_t)((i0 + r) * LSEQ + l) * 128 + o] = acc[r][j];
    }
  }
}

// ---------------------------------------------------------------------------
// Instance-norm statistics (deterministic two-level tree, no atomics)
// ---------------------------------------------------------------------------
__global__ void k_chstats(const float* __restrict__ buf,
                          float* __restrict__ psum, float* __restrict__ psq) {
  const int i = blockIdx.x;       // 0..255 rows
  const int o = threadIdx.x;      // 0..127 channels
  float s = 0.f, s2 = 0.f;
  for (int l = 0; l < LSEQ; ++l) {
    const float v = buf[(size_t)(i * LSEQ + l) * 128 + o];
    s += v; s2 += v * v;
  }
  psum[i * 128 + o] = s;
  psq[i * 128 + o] = s2;
}

__global__ void k_finstats(const float* __restrict__ psum, const float* __restrict__ psq,
                           float* __restrict__ mean, float* __restrict__ rstd) {
  const int o = threadIdx.x;      // 128 threads
  float s = 0.f, s2 = 0.f;
  for (int i = 0; i < LSEQ; ++i) { s += psum[i * 128 + o]; s2 += psq[i * 128 + o]; }
  const float m = s * (1.f / 65536.f);
  const float v = s2 * (1.f / 65536.f) - m * m;
  mean[o] = m;
  rstd[o] = rsqrtf(v + 1e-6f);
}

// norm (+ optional residual) + ELU; optional f32/f16 outputs
__global__ void k_norm_act(const float* __restrict__ pre, const float* __restrict__ mean,
                           const float* __restrict__ rstd, const float* __restrict__ g,
                           const float* __restrict__ beta, const float* __restrict__ resid,
                           float* __restrict__ out32, half_t* __restrict__ out16) {
  const size_t idx = (size_t)blockIdx.x * 256 + threadIdx.x;  // 65536*128
  const int o = (int)(idx & 127);
  float v = (pre[idx] - mean[o]) * rstd[o] * g[o] + beta[o];
  if (resid) v += resid[idx];
  v = v > 0.f ? v : (__expf(v) - 1.f);
  if (out32) out32[idx] = v;
  if (out16) out16[idx] = (half_t)v;
}

// ---------------------------------------------------------------------------
extern "C" void kernel_launch(void* const* d_in, const int* in_sizes, int n_in,
                              void* d_out, int out_size, void* d_ws, size_t ws_size,
                              hipStream_t stream) {
  const float* msa      = (const float*)d_in[0];
  const float* pair_og  = (const float*)d_in[1];
  const float* W1       = (const float*)d_in[2];
  const float* b1       = (const float*)d_in[3];
  const float* g_ln     = (const float*)d_in[4];
  const float* beta_ln  = (const float*)d_in[5];
  const float* W2       = (const float*)d_in[6];
  const float* b2       = (const float*)d_in[7];
  const float* Wc0      = (const float*)d_in[8];
  const float* g0       = (const float*)d_in[9];
  const float* beta0    = (const float*)d_in[10];
  const float* Wr1      = (const float*)d_in[11];
  const float* g1       = (const float*)d_in[12];
  const float* beta1    = (const float*)d_in[13];
  const float* Wr2      = (const float*)d_in[14];
  const float* g2       = (const float*)d_in[15];
  const float* beta2    = (const float*)d_in[16];
  float* out = (float*)d_out;                        // [256][256][128] f32 NHWC

  char* ws = (char*)d_ws;
  size_t off = 0;
  auto alloc = [&](size_t bytes) { char* p = ws + off; off = (off + bytes + 255) & ~(size_t)255; return p; };
  half_t* xdT   = (half_t*)alloc(8192 * 64 * 2);           // proj_1 output, K-major
  half_t* W2h   = (half_t*)alloc(128 * 1024 * 2);
  half_t* Wcath = (half_t*)alloc(128 * 256 * 2);
  half_t* W3t1  = (half_t*)alloc(9 * 128 * 128 * 2);
  half_t* W3t2  = (half_t*)alloc(9 * 128 * 128 * 2);
  float*  feat  = (float*)alloc(256 * 128 * 4);
  float*  Lc    = (float*)alloc(256 * 128 * 4);
  float*  Rc    = (float*)alloc(256 * 128 * 4);
  float*  psum  = (float*)alloc(256 * 128 * 4);
  float*  psq   = (float*)alloc(256 * 128 * 4);
  float*  meanb = (float*)alloc(128 * 4);
  float*  rstdb = (float*)alloc(128 * 4);
  half_t* cat   = (half_t*)alloc((size_t)65536 * 256 * 2); // pair_orig++pair2 f16; reused as h f16
  half_t* xf16  = (half_t*)alloc((size_t)65536 * 128 * 2);
  float*  P1    = (float*)alloc((size_t)65536 * 128 * 4);  // pre-norm scratch
  float*  P2    = (float*)alloc((size_t)65536 * 128 * 4);  // x (post IN+ELU) f32
  (void)ws_size; (void)n_in; (void)in_sizes; (void)out_size;

  const dim3 gPair(8, 256);    // (32-px l-block, i)
  const dim3 gTile(16, 256);   // (16-px l-block, i)
  const dim3 gConv3(16, 32);   // (16-px l-block, 8-row i-block)
  // weights + small precompute
  k_prep<<<576, 256, 0, stream>>>(W2, Wc0, Wr1, Wr2, W2h, Wcath, W3t1, W3t2);
  k_xdown<<<2048, 256, 0, stream>>>(msa, W1, b1, xdT);
  k_feat<<<128, 256, 0, stream>>>(msa, feat);
  k_lr<<<128, 256, 0, stream>>>(feat, Wc0, Lc, Rc);
  k_cvt_po<<<32768, 256, 0, stream>>>(pair_og, cat);
  // outer product + LN + proj_2 (fused, WMMA)
  k_pair<<<gPair, 256, 0, stream>>>(xdT, g_ln, beta_ln, W2h, b2, cat);
  // 1x1 conv 512->128 (WMMA K=256 + rank-1 terms)
  k_conv1<<<gTile, 256, 0, stream>>>(cat, Wcath, Lc, Rc, P1);
  // IN + ELU -> x (f32 + f16)
  k_chstats<<<256, 128, 0, stream>>>(P1, psum, psq);
  k_finstats<<<1, 128, 0, stream>>>(psum, psq, meanb, rstdb);
  k_norm_act<<<32768, 256, 0, stream>>>(P1, meanb, rstdb, g0, beta0, nullptr, P2, xf16);
  // ResBlock conv1 (3x3, WMMA implicit GEMM, async-LDS halo)
  k_conv3<<<gConv3, 256, 0, stream>>>(xf16, W3t1, P1);
  k_chstats<<<256, 128, 0, stream>>>(P1, psum, psq);
  k_finstats<<<1, 128, 0, stream>>>(psum, psq, meanb, rstdb);
  k_norm_act<<<32768, 256, 0, stream>>>(P1, meanb, rstdb, g1, beta1, nullptr, nullptr, cat /* h f16 */);
  // ResBlock conv2
  k_conv3<<<gConv3, 256, 0, stream>>>(cat, W3t2, P1);
  k_chstats<<<256, 128, 0, stream>>>(P1, psum, psq);
  k_finstats<<<1, 128, 0, stream>>>(psum, psq, meanb, rstdb);
  // final: elu(x + IN(h2)) -> d_out [L,L,128] f32
  k_norm_act<<<32768, 256, 0, stream>>>(P1, meanb, rstdb, g2, beta2, P2, out, nullptr);
}
